// HierarchicalResidualQuantizer_82978768159586
// MI455X (gfx1250) — matrix operand order repaired and checked by reading, not compiled, not run on hardware
//
#include <hip/hip_runtime.h>

typedef __attribute__((ext_vector_type(16))) _Float16 v16h;
typedef __attribute__((ext_vector_type(8)))  float    v8f;

namespace {
constexpr int   L          = 8;
constexpr int   K          = 512;
constexpr int   KP         = K + 4;     // padded score row (bank-conflict avoidance)
constexpr int   D          = 64;
constexpr int   TILE       = 16;
constexpr int   NROWS      = 16384;     // 16*32*32
constexpr int   ROWS_PER_B = 1024;      // 32*32
constexpr float GAMMA_     = 10.0f;
constexpr float KL_W       = 0.001f;
constexpr float NORM_W     = 0.1f;
constexpr float NORM_SC    = 1.5f;

// output layout: [qsum (B,D,H,W) | indices (B,H,W,L) | loss (B) | all_probs (N,L,K)]
constexpr size_t OFF_IDX   = 1048576;               // 16*64*32*32
constexpr size_t OFF_LOSS  = OFF_IDX + 131072;      // + 16*32*32*8
constexpr size_t OFF_PROBS = OFF_LOSS + 16;

struct Smem {
  float scores[TILE][KP];    // ~33 KB
  float z[TILE][D];          // 4 KB
  float qsum[TILE][D];       // 4 KB
  float rnorm2[TILE];
  float rowMax[TILE];
  float rowInvS1[TILE];
  int   idx[TILE];
  float qnorm[L][TILE];
  float loss[TILE];
};
} // namespace

__global__ void hrq_zero_loss(float* out_loss) {
  out_loss[threadIdx.x] = 0.0f;
}

__global__ __launch_bounds__(32) void hrq_main(const float* __restrict__ zg,
                                               const float* __restrict__ emb,
                                               const int*   __restrict__ epoch_p,
                                               float*       __restrict__ out) {
  __shared__ Smem s;
  const int lane = threadIdx.x;        // 0..31 (wave32)
  const int m    = lane & 15;          // row / column within 16x16 tile
  const int half = lane >> 4;          // 0/1
  const int tile = blockIdx.x;
  const int row0 = tile * TILE;
  const float epoch = (float)(*epoch_p);

  float* out_qsum  = out;
  float* out_idx   = out + OFF_IDX;
  float* out_loss  = out + OFF_LOSS;
  float* out_probs = out + OFF_PROBS;

  // ---- stage z tile (coalesced), zero qsum / loss ----
  {
    const float4* src = (const float4*)(zg + (size_t)row0 * D);
    float4* zd = (float4*)&s.z[0][0];
    float4* qd = (float4*)&s.qsum[0][0];
#pragma unroll
    for (int i = 0; i < (TILE * D / 4) / 32; ++i) {
      zd[i * 32 + lane] = src[i * 32 + lane];
      qd[i * 32 + lane] = make_float4(0.f, 0.f, 0.f, 0.f);
    }
    if (lane < TILE) s.loss[lane] = 0.0f;
  }
  __syncthreads();

  for (int h = 0; h < L; ++h) {
    const float temp = fmaxf(__expf(-epoch / (GAMMA_ * __powf(1.5f, (float)h))), 0.5f);
    const float invt = 1.0f / temp;

    // ---- residual row norms: lane (m,half) covers 32 dims of row m ----
    {
      float acc = 0.f;
#pragma unroll
      for (int dd = 0; dd < 32; ++dd) {
        int d = half * 32 + dd;
        float r = s.z[m][d] - s.qsum[m][d];
        acc += r * r;
      }
      acc += __shfl_xor(acc, 16, 32);
      if (half == 0) s.rnorm2[m] = acc;
    }
    __syncthreads();

    // ---- build A fragments (ISA 16-bit A 16x32 layout), K-slices [0,32) / [32,64) ----
    v16h a0, a1;
#pragma unroll
    for (int v = 0; v < 8; ++v) {
      int kb = 2 * v + (v >= 4 ? 8 : 0) + 8 * half;
      a0[2 * v]     = (_Float16)(s.z[m][kb]          - s.qsum[m][kb]);
      a0[2 * v + 1] = (_Float16)(s.z[m][kb + 1]      - s.qsum[m][kb + 1]);
      a1[2 * v]     = (_Float16)(s.z[m][32 + kb]     - s.qsum[m][32 + kb]);
      a1[2 * v + 1] = (_Float16)(s.z[m][32 + kb + 1] - s.qsum[m][32 + kb + 1]);
    }

    // hoist |r|^2 for this lane's 8 output rows into registers (keeps LDS out of
    // the inner-loop epilogue)
    float rn[8];
#pragma unroll
    for (int j = 0; j < 8; ++j) rn[j] = s.rnorm2[j + 8 * half];

    // ---- score all 512 codes, 16 per WMMA tile: d = 2 r.w - |r|^2 - |w|^2 ----
    for (int ct = 0; ct < 32; ++ct) {
      const int code = ct * 16 + m;   // this lane's B column
      const float* wrow = emb + ((size_t)h * K + code) * D + half * 8;
      // fragment 0: dims [8*half, 8*half+8) and [16+8*half, 24+8*half);  fragment 1: +32
      float4 A0 = ((const float4*)wrow)[0];
      float4 A1 = ((const float4*)wrow)[1];
      float4 A2 = ((const float4*)wrow)[4];
      float4 A3 = ((const float4*)wrow)[5];
      float4 B0 = ((const float4*)(wrow + 32))[0];
      float4 B1 = ((const float4*)(wrow + 32))[1];
      float4 B2 = ((const float4*)(wrow + 32))[4];
      float4 B3 = ((const float4*)(wrow + 32))[5];
      float e0[16] = {A0.x, A0.y, A0.z, A0.w, A1.x, A1.y, A1.z, A1.w,
                      A2.x, A2.y, A2.z, A2.w, A3.x, A3.y, A3.z, A3.w};
      float e1[16] = {B0.x, B0.y, B0.z, B0.w, B1.x, B1.y, B1.z, B1.w,
                      B2.x, B2.y, B2.z, B2.w, B3.x, B3.y, B3.z, B3.w};
      float wn = 0.f;
      v16h b0, b1;
#pragma unroll
      for (int i = 0; i < 16; ++i) {
        wn += e0[i] * e0[i] + e1[i] * e1[i];
        b0[i] = (_Float16)e0[i];
        b1[i] = (_Float16)e1[i];
      }
      wn += __shfl_xor(wn, 16, 32);   // full |w|^2 for this code

      v8f c = {};
      c = __builtin_amdgcn_wmma_f32_16x16x32_f16(false, a0, false, b0,
                                                 (short)0, c, false, false);
      c = __builtin_amdgcn_wmma_f32_16x16x32_f16(false, a1, false, b1,
                                                 (short)0, c, false, false);
      float* srow = &s.scores[0][0];
#pragma unroll
      for (int j = 0; j < 8; ++j) {
        int M = j + 8 * half;         // C/D layout: vgpr j, lane-half -> row
        srow[M * KP + ct * 16 + m] = __fmaf_rn(2.0f, c[j], -rn[j]) - wn;
      }
    }
    __syncthreads();

    // ---- per-row stats: 2 lanes per row scan 256 each (float4) ----
    {
      const int r = m;
      const float* srow = s.scores[r];
      float mx = -3.4e38f, sumd = 0.f;
      int best = 0;
      for (int q = half * 64; q < half * 64 + 64; ++q) {
        float4 d4 = *(const float4*)(srow + q * 4);
        sumd += (d4.x + d4.y) + (d4.z + d4.w);
        if (d4.x > mx) { mx = d4.x; best = q * 4; }
        if (d4.y > mx) { mx = d4.y; best = q * 4 + 1; }
        if (d4.z > mx) { mx = d4.z; best = q * 4 + 2; }
        if (d4.w > mx) { mx = d4.w; best = q * 4 + 3; }
      }
      float omx  = __shfl_xor(mx, 16, 32);
      int   obst = __shfl_xor(best, 16, 32);
      sumd      += __shfl_xor(sumd, 16, 32);
      if (omx > mx || (omx == mx && obst < best)) { mx = omx; best = obst; }
      float s1 = 0.f, s2 = 0.f;
      for (int q = half * 64; q < half * 64 + 64; ++q) {
        float4 d4 = *(const float4*)(srow + q * 4);
        s1 += __expf((d4.x - mx) * invt) + __expf((d4.y - mx) * invt) +
              __expf((d4.z - mx) * invt) + __expf((d4.w - mx) * invt);
        s2 += __expf(d4.x - mx) + __expf(d4.y - mx) +
              __expf(d4.z - mx) + __expf(d4.w - mx);
      }
      s1 += __shfl_xor(s1, 16, 32);
      s2 += __shfl_xor(s2, 16, 32);
      if (half == 0) {
        s.rowMax[r]   = mx;
        s.rowInvS1[r] = 1.0f / s1;
        s.idx[r]      = best;
        float meanlogp = sumd * (1.0f / K) - (mx + __logf(s2));
        s.loss[r] += KL_W * (-__logf((float)K) - meanlogp);
        out_idx[(size_t)(row0 + r) * L + h] = (float)best;
      }
    }
    __syncthreads();

    // ---- write probs (coalesced float4, all 32 lanes per row) ----
    for (int r = 0; r < TILE; ++r) {
      const float mx  = s.rowMax[r];
      const float inv = s.rowInvS1[r];
      const float* srow = s.scores[r];
      float* dst = out_probs + ((size_t)(row0 + r) * L + h) * K;
#pragma unroll
      for (int i = 0; i < 4; ++i) {
        int k = i * 128 + lane * 4;
        float4 sc = *(const float4*)(srow + k);
        float4 p;
        p.x = __expf((sc.x - mx) * invt) * inv;
        p.y = __expf((sc.y - mx) * invt) * inv;
        p.z = __expf((sc.z - mx) * invt) * inv;
        p.w = __expf((sc.w - mx) * invt) * inv;
        *(float4*)&dst[k] = p;
      }
    }

    // ---- gather selected code rows, update quant_sum, record |q|^2 ----
    for (int r = 0; r < TILE; ++r) {
      const int best = s.idx[r];
      const float* qrow = emb + ((size_t)h * K + best) * D;
      float q0 = qrow[lane * 2], q1 = qrow[lane * 2 + 1];
      s.qsum[r][lane * 2]     += q0;
      s.qsum[r][lane * 2 + 1] += q1;
      float qn = q0 * q0 + q1 * q1;
      qn += __shfl_xor(qn, 1, 32);
      qn += __shfl_xor(qn, 2, 32);
      qn += __shfl_xor(qn, 4, 32);
      qn += __shfl_xor(qn, 8, 32);
      qn += __shfl_xor(qn, 16, 32);
      if (lane == 0) s.qnorm[h][r] = qn;
    }
    __syncthreads();
  }

  // ---- norm-ordering penalty + per-batch loss ----
  if (lane < TILE) {
    const int r = lane;
    float acc = 0.f;
#pragma unroll
    for (int h = 0; h < L - 1; ++h) {
      float upper = __fsqrt_rn(s.qnorm[h][r]);
      float lower = __fsqrt_rn(s.qnorm[h + 1][r]);
      float t = fmaxf(lower / upper * NORM_SC, 1.0f) - 1.0f;
      acc += t * t;
    }
    s.loss[r] += (acc * (1.0f / (L - 1))) * NORM_W;
  }
  __syncthreads();
  {
    float part = (lane < TILE) ? s.loss[lane] : 0.f;
    part += __shfl_xor(part, 1, 32);
    part += __shfl_xor(part, 2, 32);
    part += __shfl_xor(part, 4, 32);
    part += __shfl_xor(part, 8, 32);
    part += __shfl_xor(part, 16, 32);
    if (lane == 0)
      atomicAdd(&out_loss[row0 / ROWS_PER_B], part * (1.0f / ROWS_PER_B));
  }

  // ---- quant_sum output, NHWC -> NCHW: out[b][d][p], p contiguous over 16 ----
  {
    const int b  = row0 / ROWS_PER_B;
    const int p0 = row0 % ROWS_PER_B;
#pragma unroll
    for (int dd = 0; dd < 32; ++dd) {
      int d = half * 32 + dd;
      out_qsum[(size_t)b * (D * ROWS_PER_B) + (size_t)d * ROWS_PER_B + p0 + m] =
          s.qsum[m][d];
    }
  }
}

extern "C" void kernel_launch(void* const* d_in, const int* in_sizes, int n_in,
                              void* d_out, int out_size, void* d_ws, size_t ws_size,
                              hipStream_t stream) {
  const float* z     = (const float*)d_in[0];
  const float* emb   = (const float*)d_in[1];
  const int*   epoch = (const int*)d_in[2];
  float* out = (float*)d_out;

  hrq_zero_loss<<<1, 16, 0, stream>>>(out + OFF_LOSS);
  hrq_main<<<NROWS / TILE, 32, 0, stream>>>(z, emb, epoch, out);
}
